// MHA_50895362457886
// MI455X (gfx1250) — matrix operand order repaired and checked by reading, not compiled, not run on hardware
//
#include <hip/hip_runtime.h>
#include <hip/hip_fp16.h>
#include <stdint.h>

typedef __attribute__((ext_vector_type(16))) _Float16 v16h;
typedef __attribute__((ext_vector_type(8)))  _Float16 v8h;
typedef __attribute__((ext_vector_type(8)))  float    v8f;
typedef __attribute__((ext_vector_type(4)))  unsigned int u32x4;
typedef __attribute__((ext_vector_type(8)))  int      i32x8;
typedef __attribute__((ext_vector_type(4)))  int      i32x4;

constexpr int kB    = 2;
constexpr int kS    = 2048;
constexpr int kE    = 2048;
constexpr int kQKV  = 3072;   // 128*(16+2*4)
constexpr int kNH   = 16;
constexpr int kNHKV = 4;
constexpr int kHD   = 128;

// ---------------- WMMA helpers ----------------

__device__ __forceinline__ v8f wmma_f16(v16h a, v16h b, v8f c) {
  // D = A(16x32 f16) * B(32x16 f16) + C(16x16 f32)
  return __builtin_amdgcn_wmma_f32_16x16x32_f16(false, a, false, b, (short)0, c,
                                                false, false);
}

// A-matrix 16x32 f16 fragment: lane half hh; element e -> K=16*hh+e.
// Two 16B loads forming one contiguous 32B run per lane.
__device__ __forceinline__ v16h load_a_frag(const _Float16* row, int hh) {
  v8h lo = *(const v8h*)(row + 16 * hh);
  v8h hi = *(const v8h*)(row + 16 * hh + 8);
  v16h r;
#pragma unroll
  for (int i = 0; i < 8; ++i) { r[i] = lo[i]; r[i + 8] = hi[i]; }
  return r;
}

// 16-lane (half-wave) reductions; rows of the C layout live across one half.
__device__ __forceinline__ float rmax16(float x) {
  x = fmaxf(x, __shfl_xor(x, 1, 32));
  x = fmaxf(x, __shfl_xor(x, 2, 32));
  x = fmaxf(x, __shfl_xor(x, 4, 32));
  x = fmaxf(x, __shfl_xor(x, 8, 32));
  return x;
}
__device__ __forceinline__ float rsum16(float x) {
  x += __shfl_xor(x, 1, 32);
  x += __shfl_xor(x, 2, 32);
  x += __shfl_xor(x, 4, 32);
  x += __shfl_xor(x, 8, 32);
  return x;
}

// ---------------- Tensor Data Mover (TDM) 2D tile -> LDS ----------------
// Builds a D# per CDNA5 ISA ch.8 and issues TENSOR_LOAD_TO_LDS.
// data_size = 2 bytes (f16).  2D tile: tile_x (dim0, contiguous) x tile_y rows,
// row stride = stride_elems.  LDS layout: tile rows packed contiguously.
__device__ __forceinline__ void tdm_load_2d_f16(uint32_t lds_off,
                                                const void* gptr,
                                                uint32_t tile_x, uint32_t tile_y,
                                                uint32_t tensor_x,
                                                uint32_t tensor_y,
                                                uint64_t stride_elems) {
  uint64_t ga = (uint64_t)(uintptr_t)gptr;
  u32x4 g0;
  g0[0] = 1u;                                   // count=1 (valid), no gather
  g0[1] = lds_off;                              // D#.lds_addr
  g0[2] = (uint32_t)ga;                         // global_addr[31:0]
  g0[3] = (uint32_t)((ga >> 32) & 0x1FFFFFFu)   // global_addr[56:32]
          | (2u << 30);                         // type=2 ("image")
  i32x8 g1;
  g1[0] = (int)(1u << 16);                      // wg_mask=0, data_size=1 (2B)
  g1[1] = (int)((tensor_x & 0xFFFFu) << 16);    // tensor_dim0[15:0] @ bits63:48
  g1[2] = (int)((tensor_x >> 16) | ((tensor_y & 0xFFFFu) << 16));
  g1[3] = (int)((tensor_y >> 16) | (tile_x << 16));   // tile_dim0
  g1[4] = (int)(tile_y & 0xFFFFu);              // tile_dim1 (tile_dim2=0)
  g1[5] = (int)(uint32_t)(stride_elems & 0xFFFFFFFFu);  // dim0_stride lo
  g1[6] = (int)(uint32_t)((stride_elems >> 32) & 0xFFFFu);
  g1[7] = 0;                                    // dim1_stride=0 (2D)
  i32x4 z4 = {0, 0, 0, 0};
#if defined(__clang_major__) && __clang_major__ >= 23
  i32x8 z8 = {0, 0, 0, 0, 0, 0, 0, 0};
  __builtin_amdgcn_tensor_load_to_lds(g0, g1, z4, z4, z8, 0);
#else
  __builtin_amdgcn_tensor_load_to_lds(g0, g1, z4, z4, 0);
#endif
}

// ---------------- f32 -> f16 conversion ----------------

__global__ void cvt_f32_f16(const float* __restrict__ in,
                            _Float16* __restrict__ out, size_t n) {
  size_t i = (size_t)blockIdx.x * blockDim.x + threadIdx.x;
  if (i < n) out[i] = (_Float16)in[i];
}

// ---------------- GEMM: C[M,N] = A[M,K] * B[N,K]^T + bias[N] ----------------
// Block: 256 thr = 8 waves (4 along M x 2 along N); wave tile 32x64 ->
// block tile 128(M) x 128(N).  Per K-step (32), TDM DMAs the A-slice (128x32)
// and B-slice (128x32) into LDS (double buffered); 8 WMMAs per wave per step
// run purely out of LDS.
__global__ __launch_bounds__(256) void gemm_wmma_tdm(
    const _Float16* __restrict__ A, const _Float16* __restrict__ Bm,
    const float* __restrict__ bias, float* __restrict__ C, int M, int N,
    int K) {
  __shared__ __align__(64) _Float16 Bs[2][128 * 32];  // 2 x 8KB
  __shared__ __align__(64) _Float16 As[2][128 * 32];  // 2 x 8KB

  const int lane = threadIdx.x & 31;
  const int w    = threadIdx.x >> 5;
  const int r    = lane & 15;
  const int hh   = lane >> 4;
  const int wm   = w & 3;   // 4 waves along M (32 rows each)
  const int wn   = w >> 2;  // 2 waves along N (64 cols each)
  const int mblk = blockIdx.y * 128;
  const int nblk = blockIdx.x * 128;

  const int KSTEPS = K >> 5;

  const uint32_t ldsA[2] = {(uint32_t)(uintptr_t)&As[0][0],
                            (uint32_t)(uintptr_t)&As[1][0]};
  const uint32_t ldsB[2] = {(uint32_t)(uintptr_t)&Bs[0][0],
                            (uint32_t)(uintptr_t)&Bs[1][0]};

  // Prologue: DMA slice 0 into buffer 0 (one wave drives the TDM).
  if (w == 0) {
    tdm_load_2d_f16(ldsA[0], A + (size_t)mblk * K, 32, 128, (uint32_t)K, 128,
                    (uint64_t)K);
    tdm_load_2d_f16(ldsB[0], Bm + (size_t)nblk * K, 32, 128, (uint32_t)K, 128,
                    (uint64_t)K);
  }

  v8f acc[2][4];
#pragma unroll
  for (int mt = 0; mt < 2; ++mt)
#pragma unroll
    for (int t = 0; t < 4; ++t)
#pragma unroll
      for (int v = 0; v < 8; ++v) acc[mt][t][v] = 0.0f;

  for (int i = 0; i < KSTEPS; ++i) {
    const int k0  = i << 5;
    const int cur = i & 1;
    if (w == 0) {
      if (i + 1 < KSTEPS) {
        // Prefetch next K-slice into the other buffer, then wait for the
        // current slice (<=2 outstanding = just the two we issued).
        tdm_load_2d_f16(ldsA[cur ^ 1], A + (size_t)mblk * K + (k0 + 32), 32,
                        128, (uint32_t)K, 128, (uint64_t)K);
        tdm_load_2d_f16(ldsB[cur ^ 1], Bm + (size_t)nblk * K + (k0 + 32), 32,
                        128, (uint32_t)K, 128, (uint64_t)K);
        __builtin_amdgcn_s_wait_tensorcnt(2);
      } else {
        __builtin_amdgcn_s_wait_tensorcnt(0);
      }
    }
    __syncthreads();  // current slice visible to all waves

    const _Float16* abuf = &As[cur][0];
    const _Float16* bbuf = &Bs[cur][0];
    // Two A fragments: rows wm*32 + {0,16} + r of the 128-row slice.
    v16h af0 = load_a_frag(abuf + (wm * 32 + r) * 32, hh);
    v16h af1 = load_a_frag(abuf + (wm * 32 + 16 + r) * 32, hh);
#pragma unroll
    for (int t = 0; t < 4; ++t) {
      // B fragment: column n = nblk + wn*64 + 16t + r; K = 16*hh + e.
      const _Float16* bc = bbuf + (wn * 64 + 16 * t + r) * 32;
      v16h bf    = *(const v16h*)(bc + 16 * hh);
      acc[0][t]  = wmma_f16(af0, bf, acc[0][t]);
      acc[1][t]  = wmma_f16(af1, bf, acc[1][t]);
    }
    __syncthreads();  // everyone done reading before buffer is overwritten
  }

  const int n0 = nblk + wn * 64;
#pragma unroll
  for (int mt = 0; mt < 2; ++mt) {
    const int m0 = mblk + wm * 32 + mt * 16;
#pragma unroll
    for (int t = 0; t < 4; ++t) {
      const int   n  = n0 + 16 * t + r;
      const float bv = bias[n];
#pragma unroll
      for (int v = 0; v < 8; ++v)
        C[(size_t)(m0 + v + 8 * hh) * N + n] = acc[mt][t][v] + bv;
    }
  }
}

// ---------------- depthwise causal conv (taps=4) ----------------

__device__ __forceinline__ float conv4(const float* __restrict__ qkv, int bs,
                                       int s, int f,
                                       const float* __restrict__ cw,
                                       const float* __restrict__ cb) {
  float acc = cb[f];
#pragma unroll
  for (int j = 0; j < 4; ++j) {
    int sj = s - 3 + j;
    if (sj >= 0) acc += cw[f * 4 + j] * qkv[(size_t)(bs - 3 + j) * kQKV + f];
  }
  return acc;
}

// conv + RoPE for q and k heads; one thread per (b,s,head,d<64) pair.
__global__ void conv_rope_qk(const float* __restrict__ qkv,
                             const float* __restrict__ cw,
                             const float* __restrict__ cb,
                             _Float16* __restrict__ qh,
                             _Float16* __restrict__ kh) {
  size_t       tid   = (size_t)blockIdx.x * blockDim.x + threadIdx.x;
  const size_t total = (size_t)kB * kS * (kNH + kNHKV) * 64;
  if (tid >= total) return;
  const int d    = tid & 63;
  const int head = (tid >> 6) % (kNH + kNHKV);
  const int s    = (int)((tid / (64 * (kNH + kNHKV))) % kS);
  const int b    = (int)(tid / ((size_t)64 * (kNH + kNHKV) * kS));
  const int fb   = (head < kNH) ? head * kHD : (kNH * kHD + (head - kNH) * kHD);
  const int bs   = b * kS + s;

  float x1 = conv4(qkv, bs, s, fb + d, cw, cb);
  float x2 = conv4(qkv, bs, s, fb + d + 64, cw, cb);

  float inv = powf(10000.0f, -(float)d * (1.0f / 64.0f));
  float ang = (float)s * inv;
  float c   = cosf(ang);
  float sn  = sinf(ang);
  float o1  = x1 * c - x2 * sn;
  float o2  = x2 * c + x1 * sn;

  if (head < kNH) {
    _Float16* dst = qh + ((size_t)bs * kNH + head) * kHD;
    dst[d]        = (_Float16)o1;
    dst[d + 64]   = (_Float16)o2;
  } else {
    _Float16* dst = kh + ((size_t)bs * kNHKV + (head - kNH)) * kHD;
    dst[d]        = (_Float16)o1;
    dst[d + 64]   = (_Float16)o2;
  }
}

// conv for V, written TRANSPOSED: vT[b][hkv][d][t] so attention's P@V
// B-fragments are contiguous 32B loads along t.
__global__ void conv_v_t(const float* __restrict__ qkv,
                         const float* __restrict__ cw,
                         const float* __restrict__ cb,
                         _Float16* __restrict__ vT) {
  size_t       tid   = (size_t)blockIdx.x * blockDim.x + threadIdx.x;
  const size_t total = (size_t)kB * kS * kNHKV * kHD;
  if (tid >= total) return;
  const int d  = tid & (kHD - 1);
  const int hk = (tid / kHD) % kNHKV;
  const int s  = (int)((tid / ((size_t)kHD * kNHKV)) % kS);
  const int b  = (int)(tid / ((size_t)kHD * kNHKV * kS));
  const int f  = (kNH + kNHKV) * kHD + hk * kHD + d;  // v region
  const int bs = b * kS + s;
  vT[(((size_t)b * kNHKV + hk) * kHD + d) * kS + s] =
      (_Float16)conv4(qkv, bs, s, f, cw, cb);
}

// ---------------- flash attention (causal, GQA) ----------------
// One wave per (b, head, 16-query block). Key blocks of 32.
__global__ __launch_bounds__(128) void attn_wmma(
    const _Float16* __restrict__ qh, const _Float16* __restrict__ kh,
    const _Float16* __restrict__ vT, _Float16* __restrict__ ctx) {
  __shared__ __align__(32) _Float16 pst[4][16 * 32];

  const int lane = threadIdx.x & 31;
  const int w    = threadIdx.x >> 5;
  const int r    = lane & 15;
  const int hh   = lane >> 4;
  const int gw   = blockIdx.x * 4 + w;
  const int nqb  = kS / 16;
  const int qb   = gw % nqb;
  const int head = (gw / nqb) % kNH;
  const int b    = gw / (nqb * kNH);
  const int hkv  = head / (kNH / kNHKV);
  const int s0   = qb * 16;
  const float scale = 0.08838834764831845f;  // 1/sqrt(128)

  // preload q fragments (M rows = queries, K = head dim, 4 x K32)
  v16h            qf[4];
  const _Float16* qrow = qh + ((size_t)(b * kS + s0 + r) * kNH + head) * kHD;
#pragma unroll
  for (int kk = 0; kk < 4; ++kk) qf[kk] = load_a_frag(qrow + kk * 32, hh);

  v8f   acc[8];
  float mrun[8], lrun[8];
#pragma unroll
  for (int t = 0; t < 8; ++t) {
    mrun[t] = -3.0e38f;
    lrun[t] = 0.0f;
#pragma unroll
    for (int v = 0; v < 8; ++v) acc[t][v] = 0.0f;
  }

  _Float16*       P    = pst[w];
  const _Float16* vbas = vT + ((size_t)b * kNHKV + hkv) * kHD * kS;
  const int       nkb  = (s0 + 16 + 31) >> 5;  // causal: keys <= s0+15

  for (int kb = 0; kb < nkb; ++kb) {
    const int t0 = kb * 32;
    v8f       cL, cR;
#pragma unroll
    for (int v = 0; v < 8; ++v) { cL[v] = 0.0f; cR[v] = 0.0f; }

#pragma unroll
    for (int kk = 0; kk < 4; ++kk) {
      const _Float16* kl =
          kh + ((size_t)(b * kS + t0 + r) * kNHKV + hkv) * kHD + kk * 32;
      const _Float16* kr =
          kh + ((size_t)(b * kS + t0 + 16 + r) * kNHKV + hkv) * kHD + kk * 32;
      v16h bl = *(const v16h*)(kl + 16 * hh);
      v16h br = *(const v16h*)(kr + 16 * hh);
      cL      = wmma_f16(qf[kk], bl, cL);
      cR      = wmma_f16(qf[kk], br, cR);
    }

    float alpha[8];
#pragma unroll
    for (int v = 0; v < 8; ++v) {
      const int row = s0 + v + 8 * hh;  // C layout: M = v + 8*(lane/16)
      float pl = cL[v] * scale;
      float pr = cR[v] * scale;
      if (t0 + r > row) pl = -3.0e38f;        // causal mask, col = t0+r
      if (t0 + 16 + r > row) pr = -3.0e38f;   // col = t0+16+r
      float rm   = rmax16(fmaxf(pl, pr));
      float mnew = fmaxf(mrun[v], rm);
      pl         = __expf(pl - mnew);
      pr         = __expf(pr - mnew);
      float rs   = rsum16(pl + pr);
      alpha[v]   = __expf(mrun[v] - mnew);
      mrun[v]    = mnew;
      lrun[v]    = lrun[v] * alpha[v] + rs;
      // stage P tile (16 rows x 32 local keys), row-major
      P[(v + 8 * hh) * 32 + r]      = (_Float16)pl;
      P[(v + 8 * hh) * 32 + 16 + r] = (_Float16)pr;
    }

    // same-wave LDS RAW: DS ops are in-order; fence the compiler + counter.
    asm volatile("s_wait_dscnt 0x0" ::: "memory");
    v16h pf = load_a_frag(P + r * 32, hh);  // P as A-fragment (16x32)

#pragma unroll
    for (int dt = 0; dt < 8; ++dt) {
#pragma unroll
      for (int v = 0; v < 8; ++v) acc[dt][v] *= alpha[v];
      // V fragment from transposed V: lane holds column d = dt*16+r,
      // elements e -> key t0 + 16*hh + e (contiguous along t).
      const _Float16* vrow = vbas + (size_t)(dt * 16 + r) * kS + t0;
      v16h vf = *(const v16h*)(vrow + 16 * hh);
      acc[dt] = wmma_f16(pf, vf, acc[dt]);
    }
  }

#pragma unroll
  for (int dt = 0; dt < 8; ++dt) {
#pragma unroll
    for (int v = 0; v < 8; ++v) {
      float o = acc[dt][v] / lrun[v];
      ctx[((size_t)(b * kS + s0 + v + 8 * hh)) * (kNH * kHD) + head * kHD +
          dt * 16 + r] = (_Float16)o;
    }
  }
}

// ---------------- launcher ----------------

extern "C" void kernel_launch(void* const* d_in, const int* in_sizes, int n_in,
                              void* d_out, int out_size, void* d_ws,
                              size_t ws_size, hipStream_t stream) {
  (void)in_sizes; (void)n_in; (void)out_size; (void)ws_size;
  const float* x      = (const float*)d_in[0];
  const float* W_in   = (const float*)d_in[1];
  const float* b_in   = (const float*)d_in[2];
  const float* conv_w = (const float*)d_in[3];
  const float* conv_b = (const float*)d_in[4];
  const float* W_out  = (const float*)d_in[5];
  const float* b_out  = (const float*)d_in[6];
  float*       out    = (float*)d_out;

  char*  ws  = (char*)d_ws;
  size_t off = 0;
  auto   alloc = [&](size_t bytes) {
    void* p = ws + off;
    off     = (off + bytes + 255) & ~(size_t)255;
    return p;
  };
  const size_t M = (size_t)kB * kS;  // 4096 rows
  float*    qkv   = (float*)alloc(M * kQKV * sizeof(float));
  _Float16* xh    = (_Float16*)alloc(M * kE * 2);
  _Float16* winh  = (_Float16*)alloc((size_t)kQKV * kE * 2);
  _Float16* wouth = (_Float16*)alloc((size_t)kE * (kNH * kHD) * 2);
  _Float16* qhd   = (_Float16*)alloc(M * kNH * kHD * 2);
  _Float16* khd   = (_Float16*)alloc(M * kNHKV * kHD * 2);
  _Float16* vtd   = (_Float16*)alloc(M * kNHKV * kHD * 2);  // transposed V
  _Float16* ctxh  = (_Float16*)alloc(M * kNH * kHD * 2);

  // 1) f32 -> f16 conversions
  {
    size_t n = M * kE;
    cvt_f32_f16<<<(unsigned)((n + 255) / 256), 256, 0, stream>>>(x, xh, n);
    n = (size_t)kQKV * kE;
    cvt_f32_f16<<<(unsigned)((n + 255) / 256), 256, 0, stream>>>(W_in, winh, n);
    n = (size_t)kE * kNH * kHD;
    cvt_f32_f16<<<(unsigned)((n + 255) / 256), 256, 0, stream>>>(W_out, wouth, n);
  }

  // 2) QKV projection: qkv[M, 3072] = xh @ winh^T + b_in
  gemm_wmma_tdm<<<dim3(kQKV / 128, (unsigned)(M / 128)), 256, 0, stream>>>(
      xh, winh, b_in, qkv, (int)M, kQKV, kE);

  // 3) depthwise conv + RoPE -> q/k (f16), V transposed (f16)
  {
    size_t n = (size_t)kB * kS * (kNH + kNHKV) * 64;
    conv_rope_qk<<<(unsigned)((n + 255) / 256), 256, 0, stream>>>(
        qkv, conv_w, conv_b, qhd, khd);
    n = (size_t)kB * kS * kNHKV * kHD;
    conv_v_t<<<(unsigned)((n + 255) / 256), 256, 0, stream>>>(qkv, conv_w,
                                                              conv_b, vtd);
  }

  // 4) causal GQA flash attention -> ctx (f16)
  {
    unsigned nwaves = (unsigned)(kB * kNH * (kS / 16));  // 4096
    attn_wmma<<<nwaves / 4, 128, 0, stream>>>(qhd, khd, vtd, ctxh);
  }

  // 5) output projection: out[M, 2048] = ctx @ wouth^T + b_out
  gemm_wmma_tdm<<<dim3(kE / 128, (unsigned)(M / 128)), 256, 0, stream>>>(
      ctxh, wouth, b_out, out, (int)M, kE, kE);
}
